// GroupedQueryAttention_33809982554127
// MI455X (gfx1250) — matrix operand order repaired and checked by reading, not compiled, not run on hardware
//
#include <hip/hip_runtime.h>
#include <hip/hip_bf16.h>
#include <stdint.h>

// ---------------------------------------------------------------------------
// GQA attention layer for MI455X (gfx1250), wave32, bf16 WMMA w/ fp32 accum.
// B=1, S=2048, D=2048, H=32, HKV=8, DH=64, GROUPS=4, causal, RoPE base 1e4.
// GEMMs: double-buffered LDS tiles staged with global_load_async_to_lds_b128,
// counted s_wait_asynccnt 0x4 so tile k+1's copies overlap tile k's WMMAs.
// ---------------------------------------------------------------------------

typedef __attribute__((ext_vector_type(16))) __bf16 v16bf;
typedef __attribute__((ext_vector_type(8)))  float  v8f;

union Frag16 {
    v16bf v;
    unsigned short u[16];
};

__device__ __forceinline__ unsigned short f2bf(float f) {
    unsigned int x = __builtin_bit_cast(unsigned int, f);
    x += 0x7FFFu + ((x >> 16) & 1u);           // round-to-nearest-even
    return (unsigned short)(x >> 16);
}

// Load a 16-element bf16 fragment as two 16-byte vector loads.
__device__ __forceinline__ void load_frag16(Frag16& f,
                                            const unsigned short* p0,
                                            const unsigned short* p1) {
    *(uint4*)(&f.u[0]) = *(const uint4*)p0;
    *(uint4*)(&f.u[8]) = *(const uint4*)p1;
}

// Issue one per-lane 16B global -> LDS async copy (ASYNCcnt-tracked).
__device__ __forceinline__ void async_b128(const unsigned short* gp,
                                           unsigned short* lp) {
    unsigned lds_off = (unsigned)(uintptr_t)lp;          // LDS aperture offset
    unsigned long long gaddr = (unsigned long long)(uintptr_t)gp;
    asm volatile("global_load_async_to_lds_b128 %0, %1, off"
                 :: "v"(lds_off), "v"(gaddr) : "memory");
}

// ---------------------------------------------------------------------------
// fp32 -> bf16 conversion, 4 elements / thread
// ---------------------------------------------------------------------------
__global__ void to_bf16_kernel(const float* __restrict__ X,
                               unsigned short* __restrict__ Y, int n4) {
    int i = blockIdx.x * blockDim.x + threadIdx.x;
    if (i < n4) {
        float4 f = ((const float4*)X)[i];
        ushort4 o;
        o.x = f2bf(f.x); o.y = f2bf(f.y); o.z = f2bf(f.z); o.w = f2bf(f.w);
        ((ushort4*)Y)[i] = o;
    }
}

// fp32 [Kd, Nd] -> bf16 transposed [Nd, Kd] (k-contiguous rows for GEMM B)
__global__ void transpose_bf16_kernel(const float* __restrict__ X,
                                      unsigned short* __restrict__ Y,
                                      int Kd, int Nd) {
    int i4 = blockIdx.x * blockDim.x + threadIdx.x;   // over (Kd*Nd)/4
    if (i4 >= (Kd * Nd) / 4) return;
    int i = i4 * 4;
    int n = i / Kd, k = i % Kd;                       // k..k+3 share n
    ushort4 o;
    o.x = f2bf(X[(size_t)(k + 0) * Nd + n]);
    o.y = f2bf(X[(size_t)(k + 1) * Nd + n]);
    o.z = f2bf(X[(size_t)(k + 2) * Nd + n]);
    o.w = f2bf(X[(size_t)(k + 3) * Nd + n]);
    ((ushort4*)Y)[i4] = o;
}

// ---------------------------------------------------------------------------
// C[M,N](f32) = A[M,K](bf16,row-major) @ BT[N,K](bf16,row-major = B^T)
// block = 128 threads (4 waves), tile 64x64, K-step 32, double-buffered LDS.
// 4 async b128 issues per lane per tile; in-order async completion means
// s_wait_asynccnt 0x4 retires exactly the current tile's copies while the
// next tile's copies remain in flight behind the WMMAs.
// ---------------------------------------------------------------------------
__global__ __launch_bounds__(128)
void gemm_bf16_wmma(const unsigned short* __restrict__ A,
                    const unsigned short* __restrict__ BT,
                    float* __restrict__ C, int M, int N, int K) {
    __shared__ __align__(16) unsigned short As[2][64 * 32];   // [row][k]
    __shared__ __align__(16) unsigned short Bs[2][64 * 32];   // [col][k]
    const int tid  = threadIdx.x;
    const int wave = tid >> 5;
    const int lane = tid & 31;
    const int half = lane >> 4;
    const int l15  = lane & 15;
    const int m0 = blockIdx.y * 64;
    const int n0 = blockIdx.x * 64;

    // per-lane tile copy: 2 x A + 2 x B async b128 (4 ASYNCcnt entries)
    auto issue_tile = [&](int k0, int buf) {
        for (int i = 0; i < 2; ++i) {
            int idx = (tid + i * 128) * 8;          // bf16 element in tile
            int r = idx >> 5, c = idx & 31;         // [row/col][k] layouts
            async_b128(&A[(size_t)(m0 + r) * K + k0 + c], &As[buf][idx]);
            async_b128(&BT[(size_t)(n0 + r) * K + k0 + c], &Bs[buf][idx]);
        }
    };

    const v8f vzero = {0.f, 0.f, 0.f, 0.f, 0.f, 0.f, 0.f, 0.f};
    v8f acc[4];
    for (int t = 0; t < 4; ++t) acc[t] = vzero;

    const int nk = K / 32;
    issue_tile(0, 0);
    for (int kb = 0; kb < nk; ++kb) {
        const int buf = kb & 1;
        if (kb + 1 < nk) {
            issue_tile((kb + 1) * 32, buf ^ 1);
            asm volatile("s_wait_asynccnt 0x4" ::: "memory");  // tile kb done
        } else {
            asm volatile("s_wait_asynccnt 0x0" ::: "memory");
        }
        __syncthreads();

        Frag16 a;
        const unsigned short* arow = &As[buf][(wave * 16 + l15) * 32 + 8 * half];
        load_frag16(a, arow, arow + 16);

        for (int t = 0; t < 4; ++t) {
            Frag16 b;
            const unsigned short* brow = &Bs[buf][(t * 16 + l15) * 32 + 8 * half];
            load_frag16(b, brow, brow + 16);
            acc[t] = __builtin_amdgcn_wmma_f32_16x16x32_bf16(
                false, a.v, false, b.v, (short)0, acc[t], false, false);
        }
        __syncthreads();
    }

    for (int t = 0; t < 4; ++t)
        for (int r = 0; r < 8; ++r)
            C[(size_t)(m0 + wave * 16 + r + 8 * half) * N + n0 + t * 16 + l15] =
                acc[t][r];
}

// ---------------------------------------------------------------------------
// RoPE + head-major repack: X [S, nheads*64] f32 -> Out [nheads, S, 64] bf16
// ---------------------------------------------------------------------------
__global__ void rope_pack_kernel(const float* __restrict__ X,
                                 unsigned short* __restrict__ Out, int nheads) {
    const int S = 2048;
    const int d  = threadIdx.x;                 // 0..63
    const int hh = blockIdx.x % nheads;
    const int s  = blockIdx.x / nheads;
    const float* row = X + ((size_t)s * nheads + hh) * 64;
    float x  = row[d];
    float xr = (d < 32) ? row[d + 32] : row[d - 32];
    int idx = d & 31;
    float freq = __expf(-(float)(2 * idx) * (9.210340371976184f / 64.0f));
    float ang = (float)s * freq;
    float sn, cs;
    __sincosf(ang, &sn, &cs);
    float val = (d < 32) ? (x * cs - xr * sn) : (x * cs + xr * sn);
    Out[((size_t)hh * S + s) * 64 + d] = f2bf(val);
}

// V repack TRANSPOSED: Vf [S, 8*64] f32 -> Out [8, 64, S] bf16 (d-major) so
// PV B-fragments are key-contiguous 128-bit loads.
__global__ void v_pack_kernel(const float* __restrict__ Vf,
                              unsigned short* __restrict__ Out) {
    const int S = 2048;
    int i = blockIdx.x * blockDim.x + threadIdx.x;  // over 8*64*S
    if (i >= 8 * 64 * S) return;
    int kvh = i / (64 * S);
    int rem = i % (64 * S);
    int d = rem / S;
    int s = rem % S;
    Out[i] = f2bf(Vf[(size_t)s * 512 + kvh * 64 + d]);
}

// ---------------------------------------------------------------------------
// Flash attention (causal, GQA): one block = (64 q rows, 1 head), 4 waves.
// Q: [H, S, 64] bf16 ; K: [HKV, S, 64] bf16 ; V: [HKV, 64, S] bf16 (d-major);
// Out: [S, H*64] bf16.
// ---------------------------------------------------------------------------
__global__ __launch_bounds__(128)
void flash_gqa_wmma(const unsigned short* __restrict__ Qb,
                    const unsigned short* __restrict__ Kb,
                    const unsigned short* __restrict__ Vb,
                    unsigned short* __restrict__ Ob) {
    // Per-wave P buffer stored in A-fragment order: [wave][lane(32)][e(16)]
    __shared__ __align__(16) unsigned short pbuf[4][32 * 16];
    const int S = 2048;
    const int tid  = threadIdx.x;
    const int wave = tid >> 5;
    const int lane = tid & 31;
    const int half = lane >> 4;
    const int l15  = lane & 15;
    const int qb  = blockIdx.x * 64;
    const int h   = blockIdx.y;
    const int kvh = h >> 2;                       // GROUPS = 4
    const int qrow0 = qb + wave * 16;
    const v8f vzero = {0.f, 0.f, 0.f, 0.f, 0.f, 0.f, 0.f, 0.f};

    // Q fragments, register-resident (DH=64 -> 2 K-steps), 4 x b128 loads
    Frag16 aq0, aq1;
    {
        const unsigned short* qrow = Qb + ((size_t)h * S + qrow0 + l15) * 64;
        load_frag16(aq0, qrow + 8 * half, qrow + 16 + 8 * half);
        load_frag16(aq1, qrow + 32 + 8 * half, qrow + 48 + 8 * half);
    }

    v8f acc[4];
    for (int t = 0; t < 4; ++t) acc[t] = vzero;
    float mrow[8], lrow[8];
    for (int r = 0; r < 8; ++r) { mrow[r] = -3.0e38f; lrow[r] = 0.f; }

    const int nb = (qrow0 + 16 + 31) >> 5;        // causal trip count per wave
    for (int blk = 0; blk < nb; ++blk) {
        const int k0 = blk * 32;

        // --- S = (Q K^T) * scale : two 16x16 fragments over 32 keys ---
        v8f sfrag[2];
        for (int c = 0; c < 2; ++c) {
            Frag16 bk0, bk1;
            const unsigned short* krow =
                Kb + ((size_t)kvh * S + k0 + 16 * c + l15) * 64 + 8 * half;
            load_frag16(bk0, krow, krow + 16);
            load_frag16(bk1, krow + 32, krow + 48);
            v8f s = vzero;
            s = __builtin_amdgcn_wmma_f32_16x16x32_bf16(
                false, aq0.v, false, bk0.v, (short)0, s, false, false);
            s = __builtin_amdgcn_wmma_f32_16x16x32_bf16(
                false, aq1.v, false, bk1.v, (short)0, s, false, false);
            sfrag[c] = s;
        }

        // --- online softmax (rows live in VGPR index r; reduce over 16 lanes)
        float corr[8], p0[8], p1[8];
        for (int r = 0; r < 8; ++r) {
            const int i = qrow0 + r + 8 * half;
            float s0 = sfrag[0][r] * 0.125f;      // 1/sqrt(64)
            float s1 = sfrag[1][r] * 0.125f;
            if (k0 + l15 > i)      s0 = -3.0e38f; // causal mask
            if (k0 + 16 + l15 > i) s1 = -3.0e38f;
            float loc = fmaxf(s0, s1);
            loc = fmaxf(loc, __shfl_xor(loc, 1, 32));
            loc = fmaxf(loc, __shfl_xor(loc, 2, 32));
            loc = fmaxf(loc, __shfl_xor(loc, 4, 32));
            loc = fmaxf(loc, __shfl_xor(loc, 8, 32));
            float mnew = fmaxf(mrow[r], loc);
            corr[r] = __expf(mrow[r] - mnew);
            p0[r] = __expf(s0 - mnew);
            p1[r] = __expf(s1 - mnew);
            float rs = p0[r] + p1[r];
            rs += __shfl_xor(rs, 1, 32);
            rs += __shfl_xor(rs, 2, 32);
            rs += __shfl_xor(rs, 4, 32);
            rs += __shfl_xor(rs, 8, 32);
            lrow[r] = lrow[r] * corr[r] + rs;
            mrow[r] = mnew;
        }

        // --- rescale running O ---
        for (int t = 0; t < 4; ++t)
            for (int r = 0; r < 8; ++r)
                acc[t][r] = acc[t][r] * corr[r];

        // --- scatter P into LDS directly in A-fragment order ---
        // (row,key) -> lane' = (row&15) + 16*((key>>3)&1),
        //              e'    = (key&7) + 8*(key>>4)
        for (int r = 0; r < 8; ++r) {
            const int base = ((r + 8 * half) + 16 * (l15 >> 3)) * 16 + (l15 & 7);
            pbuf[wave][base]     = f2bf(p0[r]);   // key = l15      (c=0)
            pbuf[wave][base + 8] = f2bf(p1[r]);   // key = 16 + l15 (c=1)
        }
        asm volatile("s_wait_dscnt 0x0" ::: "memory");
        Frag16 ap;
        {
            const unsigned short* pp = &pbuf[wave][lane * 16];
            load_frag16(ap, pp, pp + 8);
        }

        // --- O += P @ V : V is d-major so B-fragments are key-contiguous ---
        for (int t = 0; t < 4; ++t) {
            Frag16 bv;
            const unsigned short* vrow =
                Vb + ((size_t)kvh * 64 + t * 16 + l15) * S + k0 + 8 * half;
            load_frag16(bv, vrow, vrow + 16);
            acc[t] = __builtin_amdgcn_wmma_f32_16x16x32_bf16(
                false, ap.v, false, bv.v, (short)0, acc[t], false, false);
        }
    }

    // --- epilogue: normalize, write bf16 [S, H*64] for the Wo GEMM ---
    for (int t = 0; t < 4; ++t)
        for (int r = 0; r < 8; ++r) {
            const int row = qrow0 + r + 8 * half;
            Ob[(size_t)row * 2048 + h * 64 + t * 16 + l15] =
                f2bf(acc[t][r] / lrow[r]);
        }
}

// ---------------------------------------------------------------------------
// launch
// ---------------------------------------------------------------------------
extern "C" void kernel_launch(void* const* d_in, const int* in_sizes, int n_in,
                              void* d_out, int out_size, void* d_ws, size_t ws_size,
                              hipStream_t stream) {
    (void)in_sizes; (void)n_in; (void)out_size; (void)ws_size;
    const int S = 2048, D = 2048, H = 32, HKV = 8, DH = 64;

    const float* hid = (const float*)d_in[0];
    const float* Wq  = (const float*)d_in[1];
    const float* Wk  = (const float*)d_in[2];
    const float* Wv  = (const float*)d_in[3];
    const float* Wo  = (const float*)d_in[4];
    // d_in[5] = attention_mask: exactly causal; applied analytically in-kernel.

    char* ws = (char*)d_ws;
    size_t off = 0;
    auto alloc = [&](size_t bytes) -> void* {
        void* p = ws + off;
        off = (off + bytes + 255) & ~(size_t)255;
        return p;
    };
    unsigned short* hid_bf = (unsigned short*)alloc((size_t)S * D * 2);
    unsigned short* wqT    = (unsigned short*)alloc((size_t)(H * DH) * D * 2);
    unsigned short* wkT    = (unsigned short*)alloc((size_t)(HKV * DH) * D * 2);
    unsigned short* wvT    = (unsigned short*)alloc((size_t)(HKV * DH) * D * 2);
    unsigned short* woT    = (unsigned short*)alloc((size_t)D * (H * DH) * 2);
    float*          qf     = (float*)alloc((size_t)S * H * DH * 4);
    float*          kf     = (float*)alloc((size_t)S * HKV * DH * 4);
    float*          vf     = (float*)alloc((size_t)S * HKV * DH * 4);
    unsigned short* qrot   = (unsigned short*)alloc((size_t)H * S * DH * 2);
    unsigned short* krot   = (unsigned short*)alloc((size_t)HKV * S * DH * 2);
    unsigned short* vpk    = (unsigned short*)alloc((size_t)HKV * DH * S * 2);
    unsigned short* attn   = (unsigned short*)alloc((size_t)S * H * DH * 2);

    // 1) fp32 -> bf16; weights transposed to [out_features, in_features]
    {
        int n = S * D / 4;
        to_bf16_kernel<<<(n + 255) / 256, 256, 0, stream>>>(hid, hid_bf, n);
        n = D * H * DH / 4;
        transpose_bf16_kernel<<<(n + 255) / 256, 256, 0, stream>>>(
            Wq, wqT, D, H * DH);
        n = D * HKV * DH / 4;
        transpose_bf16_kernel<<<(n + 255) / 256, 256, 0, stream>>>(
            Wk, wkT, D, HKV * DH);
        transpose_bf16_kernel<<<(n + 255) / 256, 256, 0, stream>>>(
            Wv, wvT, D, HKV * DH);
        n = H * DH * D / 4;
        transpose_bf16_kernel<<<(n + 255) / 256, 256, 0, stream>>>(
            Wo, woT, H * DH, D);
    }

    // 2) QKV projections (bf16 WMMA, fp32 out)
    {
        dim3 gq((H * DH) / 64, S / 64);
        gemm_bf16_wmma<<<gq, 128, 0, stream>>>(hid_bf, wqT, qf, S, H * DH, D);
        dim3 gk((HKV * DH) / 64, S / 64);
        gemm_bf16_wmma<<<gk, 128, 0, stream>>>(hid_bf, wkT, kf, S, HKV * DH, D);
        gemm_bf16_wmma<<<gk, 128, 0, stream>>>(hid_bf, wvT, vf, S, HKV * DH, D);
    }

    // 3) RoPE + head-major bf16 repack; V transposed to d-major
    rope_pack_kernel<<<S * H, 64, 0, stream>>>(qf, qrot, H);
    rope_pack_kernel<<<S * HKV, 64, 0, stream>>>(kf, krot, HKV);
    {
        int n = HKV * DH * S;
        v_pack_kernel<<<(n + 255) / 256, 256, 0, stream>>>(vf, vpk);
    }

    // 4) causal GQA flash attention
    {
        dim3 gf(S / 64, H);
        flash_gqa_wmma<<<gf, 128, 0, stream>>>(qrot, krot, vpk, attn);
    }

    // 5) output projection -> fp32 d_out
    {
        dim3 go(D / 64, S / 64);
        gemm_bf16_wmma<<<go, 128, 0, stream>>>(attn, woT, (float*)d_out,
                                               S, D, H * DH);
    }
}
